// CausalMultiheadSelfAttention_81398220194275
// MI455X (gfx1250) — matrix operand order repaired and checked by reading, not compiled, not run on hardware
//
#include <hip/hip_runtime.h>
#include <hip/hip_bf16.h>

typedef _Float16 f16;
typedef __attribute__((ext_vector_type(16))) _Float16 v16h;
typedef __attribute__((ext_vector_type(8)))  float    v8f;

#define B_  2
#define S_  2048
#define D_  1024
#define H_  16
#define DK_ 64

__device__ __forceinline__ v8f zero8() {
    v8f z = {0.f, 0.f, 0.f, 0.f, 0.f, 0.f, 0.f, 0.f};
    return z;
}

// pack 8 f32 -> 8 f16 in one uint4 (single ds_store_b128)
__device__ __forceinline__ uint4 pack8(const float4 a, const float4 b) {
    union { f16 h[8]; uint4 u; } pk;
    pk.h[0] = (f16)a.x; pk.h[1] = (f16)a.y; pk.h[2] = (f16)a.z; pk.h[3] = (f16)a.w;
    pk.h[4] = (f16)b.x; pk.h[5] = (f16)b.y; pk.h[6] = (f16)b.z; pk.h[7] = (f16)b.w;
    return pk.u;
}

// pack 4 f32 -> 4 f16 in one uint2 (single ds_store_b64)
__device__ __forceinline__ uint2 pack4(float a, float b, float c, float d) {
    union { f16 h[4]; uint2 u; } pk;
    pk.h[0] = (f16)a; pk.h[1] = (f16)b; pk.h[2] = (f16)c; pk.h[3] = (f16)d;
    return pk.u;
}

__device__ __forceinline__ uint2 pack4h(f16 a, f16 b, f16 c, f16 d) {
    union { f16 h[4]; uint2 u; } pk;
    pk.h[0] = a; pk.h[1] = b; pk.h[2] = c; pk.h[3] = d;
    return pk.u;
}

__device__ __forceinline__ float fcomp(float4 v, int i) {
    return i == 0 ? v.x : i == 1 ? v.y : i == 2 ? v.z : v.w;
}

// A-matrix fragment (16xK=32, f16). ISA layout: lanes 0-15 hold row M=lane,
// K = {0..7} in VGPR0-3 and {16..23} in VGPR4-7; lanes 16-31 hold K = {8..15},{24..31}.
__device__ __forceinline__ v16h load_a_frag(const f16* tile, int ld, int lane) {
    const int m  = lane & 15;
    const int kb = (lane >> 4) * 8;
    const f16* rp = tile + m * ld;
    v16h a;
#pragma unroll
    for (int c = 0; c < 8; ++c) a[c] = rp[kb + c];
#pragma unroll
    for (int c = 0; c < 8; ++c) a[8 + c] = rp[16 + kb + c];
    return a;
}

// B-matrix fragment (K=32 x 16, f16) gathered from an LDS tile stored with the
// N index as the row (tile[n][k]): lanes 0-15 hold K={0..7},{16..23}, lanes 16-31 K={8..15},{24..31}.
__device__ __forceinline__ v16h load_bT_frag(const f16* tile, int ld, int lane) {
    const int n  = lane & 15;
    const int kb = (lane >> 4) * 8;
    const f16* rp = tile + n * ld;
    v16h b;
#pragma unroll
    for (int c = 0; c < 8; ++c) b[c] = rp[kb + c];
#pragma unroll
    for (int c = 0; c < 8; ++c) b[8 + c] = rp[16 + kb + c];
    return b;
}

__device__ __forceinline__ v8f wmma_f16(v16h a, v16h b, v8f c) {
    return __builtin_amdgcn_wmma_f32_16x16x32_f16(false, a, false, b,
                                                  (short)0, c, false, false);
}

// --- GEMM building blocks (block tile 128x128, BK=64, 8 waves of 32x64) ----

__device__ __forceinline__ void gload8(float4 v[8], const float* base) {
    const float4* gp = (const float4*)base;
#pragma unroll
    for (int g = 0; g < 8; ++g) v[g] = gp[g];
}

// B tile: thread owns 4 k-rows x 8 cols of W
__device__ __forceinline__ void gloadB(float4 v[8], const float* W, int k0, int kq, int c) {
#pragma unroll
    for (int kk = 0; kk < 4; ++kk) {
        const float4* gp = (const float4*)(W + (k0 + kq + kk) * D_ + c);
        v[2 * kk]     = gp[0];
        v[2 * kk + 1] = gp[1];
    }
}

__device__ __forceinline__ void storeA(f16 (*buf)[72], const float4 v[8], int r, int ko) {
    uint4* lp = (uint4*)&buf[r][ko];
#pragma unroll
    for (int g = 0; g < 4; ++g) lp[g] = pack8(v[2 * g], v[2 * g + 1]);
}

// transpose-store: buf[col][k], 8 x ds_store_b64
__device__ __forceinline__ void storeB(f16 (*buf)[72], const float4 v[8], int kq, int no) {
#pragma unroll
    for (int i = 0; i < 8; ++i) {
        const int half = i >> 2, sub = i & 3;
        uint2 p = pack4(fcomp(v[0 + half], sub), fcomp(v[2 + half], sub),
                        fcomp(v[4 + half], sub), fcomp(v[6 + half], sub));
        *(uint2*)&buf[no + i][kq] = p;
    }
}

__device__ __forceinline__ void gemm_compute(const f16 (*la)[72], const f16 (*lb)[72],
                                             int wm, int wn, int lane, v8f acc[2][4]) {
#pragma unroll
    for (int ks = 0; ks < 2; ++ks) {
        v16h a0 = load_a_frag(&la[32 * wm +  0][32 * ks], 72, lane);
        v16h a1 = load_a_frag(&la[32 * wm + 16][32 * ks], 72, lane);
        v16h bf[4];
#pragma unroll
        for (int j = 0; j < 4; ++j)
            bf[j] = load_bT_frag(&lb[64 * wn + 16 * j][32 * ks], 72, lane);
#pragma unroll
        for (int j = 0; j < 4; ++j) {
            acc[0][j] = wmma_f16(a0, bf[j], acc[0][j]);
            acc[1][j] = wmma_f16(a1, bf[j], acc[1][j]);
        }
    }
}

// ---------------------------------------------------------------------------
// Kernel 1: QKV projection. C[4096x1024] = x[4096x1024] @ W[1024x1024],
// grid.z selects {wq,wk,wv}. Double-buffered LDS, register-buffered pipeline.
// Output f16 in [B][H][S][DK] head-major layout.
// ---------------------------------------------------------------------------
__global__ __launch_bounds__(256)
void qkv_proj_kernel(const float* __restrict__ x,
                     const float* __restrict__ wq,
                     const float* __restrict__ wk,
                     const float* __restrict__ wv,
                     f16* __restrict__ qh, f16* __restrict__ kh,
                     f16* __restrict__ vh) {
    __shared__ f16 lds_a[2][128][72];
    __shared__ f16 lds_b[2][128][72];

    const int tid  = threadIdx.x;
    const int lane = tid & 31;
    const int w    = tid >> 5;
    const int wm   = w & 3;          // 4 row-groups of 32
    const int wn   = w >> 2;         // 2 col-groups of 64
    const int row0 = blockIdx.x * 128;
    const int col0 = blockIdx.y * 128;

    const float* W;
    f16* out;
    if (blockIdx.z == 0)      { W = wq; out = qh; }
    else if (blockIdx.z == 1) { W = wk; out = kh; }
    else                      { W = wv; out = vh; }

    v8f acc[2][4];
#pragma unroll
    for (int i = 0; i < 2; ++i)
#pragma unroll
        for (int j = 0; j < 4; ++j) acc[i][j] = zero8();

    const int ar  = tid >> 1;            // A stage: row, 32-wide half
    const int ako = (tid & 1) * 32;
    const int bkq = 4 * (tid >> 4);      // B stage: 4 k-rows
    const int bno = (tid & 15) * 8;      // 8 cols

    const int NK = D_ / 64;              // 16

    {   // prologue: stage tile 0
        float4 xa[8], wb[8];
        gload8(xa, x + (row0 + ar) * D_ + ako);
        gloadB(wb, W, 0, bkq, col0 + bno);
        storeA(lds_a[0], xa, ar, ako);
        storeB(lds_b[0], wb, bkq, bno);
    }
    __syncthreads();

    for (int kt = 0; kt < NK; ++kt) {
        const int cur  = kt & 1;
        const bool more = (kt + 1) < NK;
        float4 xa[8], wb[8];
        if (more) {
            const int kn = (kt + 1) * 64;
            gload8(xa, x + (row0 + ar) * D_ + kn + ako);
            gloadB(wb, W, kn, bkq, col0 + bno);
            if (kt + 2 < NK) {
                __builtin_prefetch(x + (row0 + ar) * D_ + kn + 64 + ako, 0, 3);
                __builtin_prefetch(W + (kn + 64 + bkq) * D_ + col0 + bno, 0, 3);
            }
        }
        gemm_compute(lds_a[cur], lds_b[cur], wm, wn, lane, acc);
        if (more) {
            storeA(lds_a[cur ^ 1], xa, ar, ako);
            storeB(lds_b[cur ^ 1], wb, bkq, bno);
        }
        __syncthreads();
    }

    // epilogue: C layout lane = N, VGPR r -> M = 8*(lane>>4)+r
#pragma unroll
    for (int i = 0; i < 2; ++i)
#pragma unroll
        for (int j = 0; j < 4; ++j)
#pragma unroll
            for (int r = 0; r < 8; ++r) {
                const int row = row0 + 32 * wm + 16 * i + 8 * (lane >> 4) + r;
                const int col = col0 + 64 * wn + 16 * j + (lane & 15);
                const int b   = row >> 11;          // / S_
                const int s   = row & (S_ - 1);
                const int h   = col >> 6;           // / DK_
                const int dk  = col & (DK_ - 1);
                out[((b * H_ + h) * S_ + s) * DK_ + dk] = (f16)acc[i][j][r];
            }
}

// ---------------------------------------------------------------------------
// Kernel 2: causal flash attention. Block = 128 threads = 4 waves; each wave
// owns 16 query rows of a 64-row block; one (b,h) per blockIdx.y.
// ---------------------------------------------------------------------------
__global__ __launch_bounds__(128)
void flash_attn_kernel(const f16* __restrict__ qg,
                       const f16* __restrict__ kg,
                       const f16* __restrict__ vg,
                       f16* __restrict__ og) {
    __shared__ f16 lds_q[64][72];      // [q_row][dim]
    __shared__ f16 lds_k[32][72];      // [key][dim]
    __shared__ f16 lds_v[64][40];      // transposed: [dim][key]
    __shared__ f16 lds_p[4][16][40];   // per-wave P tile, C-layout -> A-layout bridge

    const int tid  = threadIdx.x;
    const int lane = tid & 31;
    const int w    = tid >> 5;
    const int bh   = blockIdx.y;
    const int q0b  = blockIdx.x * 64;
    const int q0w  = q0b + 16 * w;
    const int b    = bh / H_;
    const int h    = bh % H_;

    const f16* qh = qg + (size_t)bh * S_ * DK_;
    const f16* kh = kg + (size_t)bh * S_ * DK_;
    const f16* vh = vg + (size_t)bh * S_ * DK_;

    {   // stage Q block 64x64 (vectorized copy, 64B per thread)
        const int r = tid >> 1;
        const int o = (tid & 1) * 32;
        const uint4* gp = (const uint4*)(qh + (q0b + r) * DK_ + o);
        uint4* lp = (uint4*)&lds_q[r][o];
#pragma unroll
        for (int g = 0; g < 4; ++g) lp[g] = gp[g];
    }
    __syncthreads();

    // loop-invariant Q A-fragments (K-dim 0..31 and 32..63)
    v16h aq0 = load_a_frag(&lds_q[16 * w][0],  72, lane);
    v16h aq1 = load_a_frag(&lds_q[16 * w][32], 72, lane);

    float rmax[8], rsum[8];
#pragma unroll
    for (int r = 0; r < 8; ++r) { rmax[r] = -1e30f; rsum[r] = 0.f; }
    v8f oacc[4];
#pragma unroll
    for (int d = 0; d < 4; ++d) oacc[d] = zero8();

    const float scale = 0.125f;        // 1/sqrt(DK)
    const int nloc  = lane & 15;
    const int rbase = 8 * (lane >> 4);
    const int kr  = tid >> 2;          // K stage: key row, 16-dim quarter
    const int ko  = (tid & 3) * 16;
    const int vkq = 4 * (tid >> 4);    // V stage: 4 keys
    const int vdo = (tid & 15) * 4;    // 4 dims

    for (int j0 = 0; j0 < q0b + 64; j0 += 32) {
        if (j0 + 32 < q0b + 64) {
            __builtin_prefetch(kh + (j0 + 32 + kr) * DK_ + ko, 0, 3);
            __builtin_prefetch(vh + (j0 + 32 + kr) * DK_ + ko, 0, 3);
        }
        {   // stage K block [32][64] (vectorized, 32B per thread)
            const uint4* gp = (const uint4*)(kh + (j0 + kr) * DK_ + ko);
            uint4* lp = (uint4*)&lds_k[kr][ko];
            lp[0] = gp[0];
            lp[1] = gp[1];
        }
        {   // stage V block transposed: lds_v[dim][key], packed b64 stores
            union { uint2 u; f16 hh[4]; } gv[4];
#pragma unroll
            for (int kk = 0; kk < 4; ++kk)
                gv[kk].u = *(const uint2*)(vh + (j0 + vkq + kk) * DK_ + vdo);
#pragma unroll
            for (int i = 0; i < 4; ++i)
                *(uint2*)&lds_v[vdo + i][vkq] =
                    pack4h(gv[0].hh[i], gv[1].hh[i], gv[2].hh[i], gv[3].hh[i]);
        }
        __syncthreads();

        // scores: 16 q rows x 32 keys, K-dim = 64 in two WMMA steps
        v8f sc0 = zero8(), sc1 = zero8();
        {
            v16h bkf[4];
            bkf[0] = load_bT_frag(&lds_k[ 0][ 0], 72, lane);
            bkf[1] = load_bT_frag(&lds_k[ 0][32], 72, lane);
            bkf[2] = load_bT_frag(&lds_k[16][ 0], 72, lane);
            bkf[3] = load_bT_frag(&lds_k[16][32], 72, lane);
            sc0 = wmma_f16(aq0, bkf[0], sc0);
            sc0 = wmma_f16(aq1, bkf[1], sc0);
            sc1 = wmma_f16(aq0, bkf[2], sc1);
            sc1 = wmma_f16(aq1, bkf[3], sc1);
        }

        // scale + causal mask + current block row-max
        float cmax[8];
#pragma unroll
        for (int r = 0; r < 8; ++r) {
            const int qrow = q0w + rbase + r;
            float v0 = sc0[r] * scale;
            float v1 = sc1[r] * scale;
            if (j0 + nloc      > qrow) v0 = -1e30f;
            if (j0 + 16 + nloc > qrow) v1 = -1e30f;
            sc0[r] = v0; sc1[r] = v1;
            cmax[r] = fmaxf(v0, v1);
        }
#pragma unroll
        for (int off = 1; off < 16; off <<= 1)
#pragma unroll
            for (int r = 0; r < 8; ++r)
                cmax[r] = fmaxf(cmax[r], __shfl_xor(cmax[r], off, 32));

        // online softmax rescale
        float fac[8];
#pragma unroll
        for (int r = 0; r < 8; ++r) {
            const float nm = fmaxf(rmax[r], cmax[r]);
            fac[r]  = __expf(rmax[r] - nm);
            rmax[r] = nm;
            rsum[r] *= fac[r];
        }
#pragma unroll
        for (int d = 0; d < 4; ++d)
#pragma unroll
            for (int r = 0; r < 8; ++r) oacc[d][r] *= fac[r];

        // exponentiate, accumulate row sums, spill P to per-wave LDS tile
        float psum[8];
#pragma unroll
        for (int r = 0; r < 8; ++r) {
            const float p0 = __expf(sc0[r] - rmax[r]);
            const float p1 = __expf(sc1[r] - rmax[r]);
            psum[r] = p0 + p1;
            lds_p[w][rbase + r][nloc]      = (f16)p0;
            lds_p[w][rbase + r][16 + nloc] = (f16)p1;
        }
#pragma unroll
        for (int off = 1; off < 16; off <<= 1)
#pragma unroll
            for (int r = 0; r < 8; ++r)
                psum[r] += __shfl_xor(psum[r], off, 32);
#pragma unroll
        for (int r = 0; r < 8; ++r) rsum[r] += psum[r];

        // wave-private LDS RAW: make the P stores visible before the gather
        asm volatile("s_wait_dscnt 0" ::: "memory");

        // O += P(16x32) x V(32x64)
        v16h pa = load_a_frag(&lds_p[w][0][0], 40, lane);
        v16h bv[4];
#pragma unroll
        for (int d = 0; d < 4; ++d)
            bv[d] = load_bT_frag(&lds_v[16 * d][0], 40, lane);
#pragma unroll
        for (int d = 0; d < 4; ++d)
            oacc[d] = wmma_f16(pa, bv[d], oacc[d]);
        __syncthreads();   // before next iteration restages lds_k / lds_v
    }

    // epilogue: normalize and store to [B][S][D] f16 (token-major for out proj)
#pragma unroll
    for (int d = 0; d < 4; ++d)
#pragma unroll
        for (int r = 0; r < 8; ++r) {
            const int qrow = q0w + rbase + r;
            const int dim  = 16 * d + nloc;
            const float v  = oacc[d][r] / rsum[r];
            og[((size_t)(b * S_ + qrow)) * D_ + h * DK_ + dim] = (f16)v;
        }
}

// ---------------------------------------------------------------------------
// Kernel 3: output projection. out[4096x1024] f32 = attn[4096x1024]f16 @ wo.
// Same pipelined 128x128x64 tiling; A is already f16.
// ---------------------------------------------------------------------------
__global__ __launch_bounds__(256)
void out_proj_kernel(const f16* __restrict__ A,
                     const float* __restrict__ W,
                     float* __restrict__ out) {
    __shared__ f16 lds_a[2][128][72];
    __shared__ f16 lds_b[2][128][72];

    const int tid  = threadIdx.x;
    const int lane = tid & 31;
    const int w    = tid >> 5;
    const int wm   = w & 3;
    const int wn   = w >> 2;
    const int row0 = blockIdx.x * 128;
    const int col0 = blockIdx.y * 128;

    v8f acc[2][4];
#pragma unroll
    for (int i = 0; i < 2; ++i)
#pragma unroll
        for (int j = 0; j < 4; ++j) acc[i][j] = zero8();

    const int ar  = tid >> 1;
    const int ako = (tid & 1) * 32;
    const int bkq = 4 * (tid >> 4);
    const int bno = (tid & 15) * 8;

    const int NK = D_ / 64;

    {   // prologue: stage tile 0
        uint4 av[4];
        const uint4* gp = (const uint4*)(A + (size_t)(row0 + ar) * D_ + ako);
#pragma unroll
        for (int g = 0; g < 4; ++g) av[g] = gp[g];
        float4 wb[8];
        gloadB(wb, W, 0, bkq, col0 + bno);
        uint4* lp = (uint4*)&lds_a[0][ar][ako];
#pragma unroll
        for (int g = 0; g < 4; ++g) lp[g] = av[g];
        storeB(lds_b[0], wb, bkq, bno);
    }
    __syncthreads();

    for (int kt = 0; kt < NK; ++kt) {
        const int cur  = kt & 1;
        const bool more = (kt + 1) < NK;
        uint4 av[4];
        float4 wb[8];
        if (more) {
            const int kn = (kt + 1) * 64;
            const uint4* gp = (const uint4*)(A + (size_t)(row0 + ar) * D_ + kn + ako);
#pragma unroll
            for (int g = 0; g < 4; ++g) av[g] = gp[g];
            gloadB(wb, W, kn, bkq, col0 + bno);
            if (kt + 2 < NK) {
                __builtin_prefetch(A + (size_t)(row0 + ar) * D_ + kn + 64 + ako, 0, 3);
                __builtin_prefetch(W + (kn + 64 + bkq) * D_ + col0 + bno, 0, 3);
            }
        }
        gemm_compute(lds_a[cur], lds_b[cur], wm, wn, lane, acc);
        if (more) {
            uint4* lp = (uint4*)&lds_a[cur ^ 1][ar][ako];
#pragma unroll
            for (int g = 0; g < 4; ++g) lp[g] = av[g];
            storeB(lds_b[cur ^ 1], wb, bkq, bno);
        }
        __syncthreads();
    }

#pragma unroll
    for (int i = 0; i < 2; ++i)
#pragma unroll
        for (int j = 0; j < 4; ++j)
#pragma unroll
            for (int r = 0; r < 8; ++r) {
                const int row = row0 + 32 * wm + 16 * i + 8 * (lane >> 4) + r;
                const int col = col0 + 64 * wn + 16 * j + (lane & 15);
                out[(size_t)row * D_ + col] = acc[i][j][r];
            }
}

extern "C" void kernel_launch(void* const* d_in, const int* in_sizes, int n_in,
                              void* d_out, int out_size, void* d_ws, size_t ws_size,
                              hipStream_t stream) {
    const float* x  = (const float*)d_in[0];
    const float* wq = (const float*)d_in[1];
    const float* wk = (const float*)d_in[2];
    const float* wv = (const float*)d_in[3];
    const float* wo = (const float*)d_in[4];
    float* out = (float*)d_out;

    const size_t elems = (size_t)B_ * H_ * S_ * DK_;   // 4,194,304
    f16* qh = (f16*)d_ws;
    f16* kh = qh + elems;
    f16* vh = kh + elems;
    f16* ah = vh + elems;                              // total 32 MB of d_ws

    dim3 gProj(B_ * S_ / 128, D_ / 128, 3);            // (32, 8, 3)
    qkv_proj_kernel<<<gProj, 256, 0, stream>>>(x, wq, wk, wv, qh, kh, vh);

    dim3 gAttn(S_ / 64, B_ * H_);                      // (32, 32)
    flash_attn_kernel<<<gAttn, 128, 0, stream>>>(qh, kh, vh, ah);

    dim3 gOut(B_ * S_ / 128, D_ / 128, 1);             // (32, 8)
    out_proj_kernel<<<gOut, 256, 0, stream>>>(ah, wo, out);
}
